// nmrMPNN_32109175505578
// MI455X (gfx1250) — compile-verified, matrix operand order checked by reading, not run on hardware
//
#include <hip/hip_runtime.h>
#include <hip/hip_bf16.h>
#include <math.h>
#include <stdint.h>

// ---------------------------------------------------------------------------
// MPNN (NNConv + GRU) for MI455X / gfx1250, wave32, WMMA bf16.
//
// Perf model: W_e (E x 64 x 64) dominates: stored bf16 (0.8 GB) and streamed
// 6x (4.8 GB) -> ~0.24 ms at 23.3 TB/s. W_e is written/read with NT cache
// policy so it never thrashes the 192MB L2 (h/agg/weights stay resident).
// All GEMMs use v_wmma_f32_16x16x32_bf16; B-fragments come from LDS
// pre-swizzled to fragment order (one aligned 32B read = 2x ds_load_b128).
// ---------------------------------------------------------------------------

#define NNODES 50000
#define NEDGES 100000
#define DD     64
#define EHID   128
#define PHID   512
#define NSTEPS 6

typedef __attribute__((ext_vector_type(16))) __bf16    v16bf;
typedef __attribute__((ext_vector_type(8)))  float     v8f;
typedef __attribute__((ext_vector_type(4)))  uint32_t  v4u;

// K index inside a bf16 16x32 A/B fragment, per CDNA5 ISA 7.12.2.
// For fixed half: e=0..7 -> k = half*8 + e ; e=8..15 -> k = 16 + half*8 + e-8.
__device__ __forceinline__ int frag_k(int e, int half) {
  int v = e >> 1, slot = e & 1;
  int group = v >> 2, within = v & 3;
  return group * 16 + half * 8 + within * 2 + slot;
}

// Vectorized A-fragment load: two 8-float contiguous runs per lane-half,
// loaded as float4 pairs (global_load_b128). K is always a multiple of 8 here,
// so each run is fully in-range or fully zero.
__device__ __forceinline__ v16bf load_a_frag(const float* __restrict__ X,
                                             int row0, int M, int ldx,
                                             int kbase, int K, bool relu_in) {
  int lane = (int)(threadIdx.x & 31);
  int m = lane & 15, half = lane >> 4;
  int row = row0 + m;
  bool rok = row < M;
  const float* base = X + (size_t)row * ldx + kbase + half * 8;
  bool ok1 = rok && (kbase + half * 8) < K;
  bool ok2 = rok && (kbase + 16 + half * 8) < K;
  float4 z = make_float4(0.f, 0.f, 0.f, 0.f);
  float4 a0 = ok1 ? ((const float4*)base)[0] : z;
  float4 a1 = ok1 ? ((const float4*)base)[1] : z;
  float4 a2 = ok2 ? ((const float4*)(base + 16))[0] : z;
  float4 a3 = ok2 ? ((const float4*)(base + 16))[1] : z;
  float f[16] = {a0.x, a0.y, a0.z, a0.w, a1.x, a1.y, a1.z, a1.w,
                 a2.x, a2.y, a2.z, a2.w, a3.x, a3.y, a3.z, a3.w};
  v16bf a;
#pragma unroll
  for (int e = 0; e < 16; ++e) {
    float v = relu_in ? fmaxf(f[e], 0.f) : f[e];
    a[e] = (__bf16)v;
  }
  return a;
}

__device__ __forceinline__ v8f wmma_bf16(v16bf a, v16bf b, v8f c) {
  return __builtin_amdgcn_wmma_f32_16x16x32_bf16(
      /*neg_a=*/false, a, /*neg_b=*/false, b,
      /*c_mod=*/(short)0, c, /*reuse_a=*/false, /*reuse_b=*/false);
}

// ---------------------------------------------------------------------------
// Generic fused GEMM: Y = act(X[MxK] @ W[KxN] + bias). W is staged per
// 32-row K-chunk into LDS *pre-swizzled* into fragment order
// [ntile][lane][elem]; each thread handles one chunk-row x 8 columns so the
// global read is 2x b128 and the swizzle store is straight-line (no exec
// masking, no 64-bit per-element address math).
// ---------------------------------------------------------------------------
#define GWAVES 8
#define GNT    64

__global__ void __launch_bounds__(256)
gemm_bias_act(const float* __restrict__ X, const float* __restrict__ W,
              const float* __restrict__ bias, float* __restrict__ Yf,
              __bf16* __restrict__ Yb, int M, int K, int N, int ldx, int relu) {
  __shared__ __align__(32) __bf16 Wswz[4][32][16];  // 4KB, fragment-ready

  int wave = (int)(threadIdx.x >> 5);
  int lane = (int)(threadIdx.x & 31);
  int row0 = ((int)blockIdx.x * GWAVES + wave) * 16;
  int ncol0 = (int)blockIdx.y * GNT;

  // Fill-role decomposition: this thread owns chunk-row kk, columns nb..nb+7.
  int kk = (int)(threadIdx.x >> 3);        // 0..31
  int nb = ((int)threadIdx.x & 7) * 8;     // 0,8,...,56
  int ghalf = (kk >> 3) & 1;
  int gee = (kk >> 4) * 8 + (kk & 7);      // fragment element index for kk
  bool fullN = (ncol0 + GNT <= N);

  v8f acc[GNT / 16] = {};

  for (int kbase = 0; kbase < K; kbase += 32) {
    int k = kbase + kk;
    if (fullN && k < K) {
      const float* wrow = W + (size_t)k * N + ncol0 + nb;
      float4 w0 = ((const float4*)wrow)[0];
      float4 w1 = ((const float4*)wrow)[1];
      float wv[8] = {w0.x, w0.y, w0.z, w0.w, w1.x, w1.y, w1.z, w1.w};
#pragma unroll
      for (int j = 0; j < 8; ++j) {
        int nn = nb + j;
        Wswz[nn >> 4][ghalf * 16 + (nn & 15)][gee] = (__bf16)wv[j];
      }
    } else {
#pragma unroll
      for (int j = 0; j < 8; ++j) {
        int n = ncol0 + nb + j;
        int widx = min(k, K - 1) * N + min(n, N - 1);  // always-safe address
        float wv = W[widx];
        bool ok = (k < K) && (n < N);
        int nn = nb + j;
        Wswz[nn >> 4][ghalf * 16 + (nn & 15)][gee] =
            ok ? (__bf16)wv : (__bf16)0.f;
      }
    }
    __syncthreads();

    v16bf a = load_a_frag(X, row0, M, ldx, kbase, K, false);
#pragma unroll
    for (int t = 0; t < GNT / 16; ++t) {
      v16bf b = *reinterpret_cast<const v16bf*>(&Wswz[t][lane][0]);
      acc[t] = wmma_bf16(a, b, acc[t]);
    }
    __syncthreads();
  }

  // Epilogue: bias + activation; fast path when the whole tile is in range.
  int nl = lane & 15, half = lane >> 4;
  bool fullM = (row0 + 16 <= M);
  if (fullM && fullN) {
#pragma unroll
    for (int t = 0; t < GNT / 16; ++t) {
      int col = ncol0 + t * 16 + nl;
      float bv = bias[col];
#pragma unroll
      for (int r = 0; r < 8; ++r) {
        int row = row0 + half * 8 + r;
        float v = acc[t][r] + bv;
        if (relu) v = fmaxf(v, 0.f);
        if (Yb) {
          // W_e production: non-temporal, 819MB stream must not evict L2.
          __builtin_nontemporal_store((__bf16)v, &Yb[(size_t)row * N + col]);
        } else {
          Yf[(size_t)row * N + col] = v;
        }
      }
    }
  } else {
#pragma unroll
    for (int t = 0; t < GNT / 16; ++t) {
#pragma unroll
      for (int r = 0; r < 8; ++r) {
        int row = row0 + half * 8 + r;
        int col = ncol0 + t * 16 + nl;
        if (row < M && col < N) {
          float v = acc[t][r] + bias[col];
          if (relu) v = fmaxf(v, 0.f);
          if (Yb) __builtin_nontemporal_store((__bf16)v, &Yb[(size_t)row * N + col]);
          else    Yf[(size_t)row * N + col] = v;
        }
      }
    }
  }
}

// ---------------------------------------------------------------------------
// Message + scatter: msg[e] = h[src[e]] (1x64) . W_e[e] (64x64 bf16),
// agg[dst[e]] += msg[e]. One wave per edge. Each lane streams 16B (b128, NT)
// of W_e per iteration (coalesced 512B per wave-instruction); lanes form
// 4 row-groups x 8 column-chunks; two shfl_xor stages reduce row-groups.
// ---------------------------------------------------------------------------
__global__ void __launch_bounds__(256)
message_scatter(const float* __restrict__ h, const __bf16* __restrict__ We,
                const int* __restrict__ src, const int* __restrict__ dst,
                float* __restrict__ agg, int E) {
  __shared__ float xs[8][DD];
  int wave = (int)(threadIdx.x >> 5);
  int lane = (int)(threadIdx.x & 31);
  int e = (int)blockIdx.x * 8 + wave;

  int s = (e < E) ? src[e] : 0;
  xs[wave][lane]      = (e < E) ? h[(size_t)s * DD + lane] : 0.f;
  xs[wave][lane + 32] = (e < E) ? h[(size_t)s * DD + lane + 32] : 0.f;
  __syncthreads();

  if (e < E) {
    int sub = lane >> 3;  // row within each 4-row group
    int cl  = lane & 7;   // 16B column chunk: cols 8*cl .. 8*cl+7
    const v4u* wp = reinterpret_cast<const v4u*>(We + (size_t)e * (DD * DD));
    float acc[8] = {};
#pragma unroll 4
    for (int it = 0; it < 16; ++it) {
      int d = it * 4 + sub;
      v4u w = __builtin_nontemporal_load(wp + it * 32 + lane);  // NT stream
      float xd = xs[wave][d];
      uint32_t pk[4] = {w.x, w.y, w.z, w.w};
#pragma unroll
      for (int j = 0; j < 4; ++j) {
        float w0 = __uint_as_float((pk[j] & 0xffffu) << 16);
        float w1 = __uint_as_float(pk[j] & 0xffff0000u);
        acc[2 * j]     = fmaf(xd, w0, acc[2 * j]);
        acc[2 * j + 1] = fmaf(xd, w1, acc[2 * j + 1]);
      }
    }
    // Reduce across the 4 row-groups (lanes l, l^8, l^16, l^24).
#pragma unroll
    for (int j = 0; j < 8; ++j) {
      acc[j] += __shfl_xor(acc[j], 8, 32);
      acc[j] += __shfl_xor(acc[j], 16, 32);
    }
    // Each lane commits 2 of its chunk's 8 columns: 64 balanced atomics/edge.
    float o0 = (sub == 0) ? acc[0] : (sub == 1) ? acc[2] : (sub == 2) ? acc[4] : acc[6];
    float o1 = (sub == 0) ? acc[1] : (sub == 1) ? acc[3] : (sub == 2) ? acc[5] : acc[7];
    int t = dst[e];
    int c0 = cl * 8 + sub * 2;
    atomicAdd(&agg[(size_t)t * DD + c0 + 0], o0);
    atomicAdd(&agg[(size_t)t * DD + c0 + 1], o1);
  }
}

// ---------------------------------------------------------------------------
// Fused ReLU + GRU step. x = relu(agg); gi = x@Wih^T, gh = h@Whh^T via WMMA.
// Gate weights are stored in LDS pre-swizzled to fragment order, so each
// B-fragment is a single 32B LDS read. C-layout trick: channels c, c+64,
// c+128 share (lane, element) -> r/z/n gate math is register-local.
// ---------------------------------------------------------------------------
__global__ void __launch_bounds__(256)
gru_step(const float* __restrict__ agg, float* __restrict__ h,
         const float* __restrict__ Wih, const float* __restrict__ Whh,
         const float* __restrict__ bih, const float* __restrict__ bhh, int Nn) {
  // [kc][tile][lane][elem]: B[k][n] = W[n][k], fragment-ready. 24KB each.
  __shared__ __align__(32) __bf16 Bih[2][12][32][16];
  __shared__ __align__(32) __bf16 Bhh[2][12][32][16];
  for (int idx = (int)threadIdx.x; idx < DD * 3 * DD; idx += 256) {
    int n = idx >> 6, k = idx & 63;  // Wih[n][k], linear (coalesced) read
    int kc = k >> 5, kr = k & 31;
    int group = kr >> 4, half = (kr >> 3) & 1, low3 = kr & 7;
    int ee = group * 8 + low3;       // inverse of frag_k
    int tt = n >> 4, nl = n & 15;
    int ln = half * 16 + nl;
    Bih[kc][tt][ln][ee] = (__bf16)Wih[idx];
    Bhh[kc][tt][ln][ee] = (__bf16)Whh[idx];
  }
  __syncthreads();

  int wave = (int)(threadIdx.x >> 5);
  int lane = (int)(threadIdx.x & 31);
  int row0 = ((int)blockIdx.x * GWAVES + wave) * 16;
  int nl = lane & 15, half = lane >> 4;

  v16bf ax[2], ah[2];
#pragma unroll
  for (int kc = 0; kc < 2; ++kc) {
    ax[kc] = load_a_frag(agg, row0, Nn, DD, kc * 32, DD, /*relu=*/true);
    ah[kc] = load_a_frag(h,   row0, Nn, DD, kc * 32, DD, /*relu=*/false);
  }

#pragma unroll
  for (int t = 0; t < 4; ++t) {  // channel block c = 16t + nl
    v8f ir = {}, iz = {}, in_ = {}, hr = {}, hz = {}, hn = {};
#pragma unroll
    for (int kc = 0; kc < 2; ++kc) {
#pragma unroll
      for (int g = 0; g < 3; ++g) {  // r, z, n gates -> tiles t, t+4, t+8
        v16bf bi = *reinterpret_cast<const v16bf*>(&Bih[kc][t + 4 * g][lane][0]);
        v16bf bh = *reinterpret_cast<const v16bf*>(&Bhh[kc][t + 4 * g][lane][0]);
        if (g == 0)      { ir  = wmma_bf16(ax[kc], bi, ir);  hr = wmma_bf16(ah[kc], bh, hr); }
        else if (g == 1) { iz  = wmma_bf16(ax[kc], bi, iz);  hz = wmma_bf16(ah[kc], bh, hz); }
        else             { in_ = wmma_bf16(ax[kc], bi, in_); hn = wmma_bf16(ah[kc], bh, hn); }
      }
    }
    int c = t * 16 + nl;
#pragma unroll
    for (int r = 0; r < 8; ++r) {
      int row = row0 + half * 8 + r;
      if (row < Nn) {
        float gr = (ir[r] + bih[c])      + (hr[r] + bhh[c]);
        float gz = (iz[r] + bih[DD + c]) + (hz[r] + bhh[DD + c]);
        float hnb = hn[r] + bhh[2 * DD + c];
        float rg = 1.f / (1.f + __expf(-gr));
        float zg = 1.f / (1.f + __expf(-gz));
        float ng = tanhf(in_[r] + bih[2 * DD + c] + rg * hnb);
        float hold = h[(size_t)row * DD + c];
        h[(size_t)row * DD + c] = (1.f - zg) * ng + zg * hold;
      }
    }
  }
}

// ---------------------------------------------------------------------------
// Host-side orchestration.
// ---------------------------------------------------------------------------
static void launch_gemm(const float* X, const float* W, const float* b,
                        float* Yf, __bf16* Yb, int M, int K, int N, int ldx,
                        int relu, hipStream_t stream) {
  dim3 grid((M + GWAVES * 16 - 1) / (GWAVES * 16), (N + GNT - 1) / GNT);
  gemm_bias_act<<<grid, 256, 0, stream>>>(X, W, b, Yf, Yb, M, K, N, ldx, relu);
}

extern "C" void kernel_launch(void* const* d_in, const int* in_sizes, int n_in,
                              void* d_out, int out_size, void* d_ws, size_t ws_size,
                              hipStream_t stream) {
  // setup_inputs() dict order, lists flattened element-wise:
  const float* node_feats = (const float*)d_in[0];   // [50000,64]
  const float* edge_feats = (const float*)d_in[1];   // [100000,16]
  const int*   src        = (const int*)d_in[2];
  const int*   dst        = (const int*)d_in[3];
  const float* pW[4] = {(const float*)d_in[4],  (const float*)d_in[5],
                        (const float*)d_in[6],  (const float*)d_in[7]};
  const float* pb[4] = {(const float*)d_in[8],  (const float*)d_in[9],
                        (const float*)d_in[10], (const float*)d_in[11]};
  const float* eW[4] = {(const float*)d_in[12], (const float*)d_in[13],
                        (const float*)d_in[14], (const float*)d_in[15]};
  const float* eb[4] = {(const float*)d_in[16], (const float*)d_in[17],
                        (const float*)d_in[18], (const float*)d_in[19]};
  const float* gWih = (const float*)d_in[20];        // [192,64]
  const float* gWhh = (const float*)d_in[21];
  const float* gbih = (const float*)d_in[22];
  const float* gbhh = (const float*)d_in[23];
  const float* qW[3] = {(const float*)d_in[24], (const float*)d_in[25],
                        (const float*)d_in[26]};
  const float* qb[3] = {(const float*)d_in[27], (const float*)d_in[28],
                        (const float*)d_in[29]};

  // Workspace carve-up.
  char* ws = (char*)d_ws;
  size_t off = 0;
  auto carve = [&](size_t bytes) {
    char* p = ws + off;
    off += (bytes + 255) & ~(size_t)255;
    return p;
  };
  float*  h    = (float*)carve((size_t)NNODES * DD * 4);
  float*  agg  = (float*)carve((size_t)NNODES * DD * 4);
  float*  nA   = (float*)carve((size_t)NNODES * PHID * 4);  // node ping
  float*  nB   = (float*)carve((size_t)NNODES * PHID * 4);  // node pong
  float*  eA   = (float*)carve((size_t)NEDGES * EHID * 4);  // edge ping
  float*  eB   = (float*)carve((size_t)NEDGES * EHID * 4);  // edge pong
  __bf16* We   = (__bf16*)carve((size_t)NEDGES * DD * DD * 2);

  // 1) Node projection MLP: 64 -> 64 x4, ReLU each; final result -> h.
  launch_gemm(node_feats, pW[0], pb[0], nA, nullptr, NNODES, 64, DD, 64, 1, stream);
  launch_gemm(nA,         pW[1], pb[1], nB, nullptr, NNODES, DD, DD, DD, 1, stream);
  launch_gemm(nB,         pW[2], pb[2], nA, nullptr, NNODES, DD, DD, DD, 1, stream);
  launch_gemm(nA,         pW[3], pb[3], h,  nullptr, NNODES, DD, DD, DD, 1, stream);

  // 2) Edge network: 16 -> 128 -> 128 -> 128 -> 4096 (no final ReLU), bf16 out.
  launch_gemm(edge_feats, eW[0], eb[0], eA, nullptr, NEDGES, 16,   EHID, 16,   1, stream);
  launch_gemm(eA,         eW[1], eb[1], eB, nullptr, NEDGES, EHID, EHID, EHID, 1, stream);
  launch_gemm(eB,         eW[2], eb[2], eA, nullptr, NEDGES, EHID, EHID, EHID, 1, stream);
  launch_gemm(eA,         eW[3], eb[3], nullptr, We, NEDGES, EHID, DD * DD, EHID, 0, stream);

  // 3) Message passing: 6 steps of NNConv scatter + GRU.
  for (int step = 0; step < NSTEPS; ++step) {
    hipMemsetAsync(agg, 0, (size_t)NNODES * DD * 4, stream);
    message_scatter<<<(NEDGES + 7) / 8, 256, 0, stream>>>(h, We, src, dst, agg, NEDGES);
    gru_step<<<(NNODES + GWAVES * 16 - 1) / (GWAVES * 16), 256, 0, stream>>>(
        agg, h, gWih, gWhh, gbih, gbhh, NNODES);
  }

  // 4) Predict MLP: 64 -> 512 -> 512 -> 1; final fp32 straight into d_out.
  launch_gemm(h,  qW[0], qb[0], nA, nullptr, NNODES, DD,   PHID, DD,   1, stream);
  launch_gemm(nA, qW[1], qb[1], nB, nullptr, NNODES, PHID, PHID, PHID, 1, stream);
  launch_gemm(nB, qW[2], qb[2], (float*)d_out, nullptr, NNODES, PHID, 1, PHID, 0, stream);
}
